// VoxMLP_5789615915256
// MI455X (gfx1250) — matrix operand order repaired and checked by reading, not compiled, not run on hardware
//
#include <hip/hip_runtime.h>

// ---------------------------------------------------------------------------
// VoxMLP for gfx1250 (MI455X): trilinear(value+grad) -> posenc -> f16 WMMA MLP
// -> Rodrigues rotation.  Compute-bound on the MLP (~138 GFLOP for 1M pts) so
// all matmuls use v_wmma_f32_16x16x32_f16 with f32 accumulation.
// Inline-0 C for first WMMA of each chain (bias folded into the store stage),
// both 16-row tiles of a wave processed jointly so each B-fragment LDS load
// feeds two WMMAs, weight staging via the Tensor Data Mover (6-arg builtin).
// ---------------------------------------------------------------------------

typedef _Float16 v16h __attribute__((ext_vector_type(16)));
typedef _Float16 v8h  __attribute__((ext_vector_type(8)));
typedef float    v8f  __attribute__((ext_vector_type(8)));
typedef unsigned int u32x4 __attribute__((ext_vector_type(4)));
typedef int          i32x4 __attribute__((ext_vector_type(4)));
typedef int          i32x8 __attribute__((ext_vector_type(8)));

#define NPTS    1048576
#define GRES    256
#define HSTRIDE 192          // activation row stride in halfs (128 h + 64 feat)
#define INV2D   63.75f       // 1/(2*delta), delta = 2/255

// packed f16 weight offsets (in halfs) inside d_ws, all column-major [n][k]
#define OFF_W0  0            // Kpad=64  (63 real), N=128  ->  8192
#define OFF_W1  8192         // Kpad=128,          N=128  -> 16384
#define OFF_W2  24576        // Kpad=128,          N=128  -> 16384
#define OFF_W3  40960        // Kpad=192 (191),    N=128  -> 24576
#define OFF_WO  65536        // Kpad=128,          N=16 (3 real) -> 2048
#define W_TOTAL 67584        // halfs; 135168 bytes = 16896 x 8B

// ---------------------------------------------------------------------------
// Weight repack: f32 row-major (K,N) -> f16 column-major (N,Kpad), zero padded
// ---------------------------------------------------------------------------
__global__ void vox_prep_weights(const float* __restrict__ w0, const float* __restrict__ w1,
                                 const float* __restrict__ w2, const float* __restrict__ w3,
                                 const float* __restrict__ wo, _Float16* __restrict__ out)
{
  int i = blockIdx.x * 256 + threadIdx.x;
  if (i >= W_TOTAL) return;
  float v = 0.0f;
  if (i < OFF_W1)      { int j = i;          int n = j / 64,  k = j % 64;  if (k < 63)  v = w0[k * 128 + n]; }
  else if (i < OFF_W2) { int j = i - OFF_W1; int n = j / 128, k = j % 128;              v = w1[k * 128 + n]; }
  else if (i < OFF_W3) { int j = i - OFF_W2; int n = j / 128, k = j % 128;              v = w2[k * 128 + n]; }
  else if (i < OFF_WO) { int j = i - OFF_W3; int n = j / 192, k = j % 192; if (k < 191) v = w3[k * 128 + n]; }
  else                 { int j = i - OFF_WO; int n = j / 128, k = j % 128; if (n < 3)   v = wo[k * 3 + n]; }
  out[i] = (_Float16)v;
}

// ---------------------------------------------------------------------------
// grid helpers
// ---------------------------------------------------------------------------
__device__ __forceinline__ float gval(const float* __restrict__ g, int i, int j, int k) {
  return g[(i << 16) + (j << 8) + k];
}
__device__ __forceinline__ float4 corner4(const float* __restrict__ g, int i, int j, int k) {
  float4 r;
  r.x = gval(g, i, j, k);
  r.y = (gval(g, (i < GRES-1) ? i+1 : GRES-1, j, k) - gval(g, (i > 0) ? i-1 : 0, j, k)) * INV2D;
  r.z = (gval(g, i, (j < GRES-1) ? j+1 : GRES-1, k) - gval(g, i, (j > 0) ? j-1 : 0, k)) * INV2D;
  r.w = (gval(g, i, j, (k < GRES-1) ? k+1 : GRES-1) - gval(g, i, j, (k > 0) ? k-1 : 0)) * INV2D;
  return r;
}
__device__ __forceinline__ float4 lerp4(float4 a, float4 b, float w) {
  float4 r;
  r.x = a.x * (1.0f - w) + b.x * w;
  r.y = a.y * (1.0f - w) + b.y * w;
  r.z = a.z * (1.0f - w) + b.z * w;
  r.w = a.w * (1.0f - w) + b.w * w;
  return r;
}

// ---------------------------------------------------------------------------
// WMMA layer helpers (layouts per CDNA5 ISA §7.12.2, wave32)
//   A 16x32 f16 : lane holds row m=lane&15; VGPR0..7 = K kb+hh*16..+15
//                 (contiguous k order, same as B; not the ISA table order)  (hh = lane>>4)
//   B 32x16 f16 : lane holds col n=lane&15; 16 consecutive K halfs at
//                 kb + hh*16  (column-major weights make this contiguous)
//   D 16x16 f32 : VGPR r = row r + 8*hh, col = lane&15
// ---------------------------------------------------------------------------
__device__ __forceinline__ v16h load_afrag(const _Float16* __restrict__ hrow, int kb, int hh) {
  v8h lo = *(const v8h*)(hrow + kb + hh * 16);
  v8h hi = *(const v8h*)(hrow + kb + hh * 16 + 8);
  return __builtin_shufflevector(lo, hi, 0,1,2,3,4,5,6,7,8,9,10,11,12,13,14,15);
}

// One layer for BOTH 16-row tiles of a wave: each B fragment load feeds two
// WMMAs.  First k-step uses an inline-zero C (bias is added at store time).
template<int KSTEPS, int NT>
__device__ __forceinline__ void joint_layer(const _Float16* __restrict__ hrow0,
                                            const _Float16* __restrict__ hrow1,
                                            int colBase,
                                            const _Float16* __restrict__ W, int Kpad,
                                            int lane, v8f* acc0, v8f* acc1)
{
  const int m  = lane & 15;
  const int hh = lane >> 4;
  const v8f Z = {0.f, 0.f, 0.f, 0.f, 0.f, 0.f, 0.f, 0.f};
#pragma unroll
  for (int s = 0; s < KSTEPS; ++s) {
    const int kb = colBase + 32 * s;
    v16h A0 = load_afrag(hrow0, kb, hh);
    v16h A1 = load_afrag(hrow1, kb, hh);
#pragma unroll
    for (int t = 0; t < NT; ++t) {
      v16h Bm = *(const v16h*)(W + (t * 16 + m) * Kpad + 32 * s + hh * 16);
      if (s == 0) {
        acc0[t] = __builtin_amdgcn_wmma_f32_16x16x32_f16(false, A0, false, Bm, (short)0, Z,       false, false);
        acc1[t] = __builtin_amdgcn_wmma_f32_16x16x32_f16(false, A1, false, Bm, (short)0, Z,       false, false);
      } else {
        acc0[t] = __builtin_amdgcn_wmma_f32_16x16x32_f16(false, A0, false, Bm, (short)0, acc0[t], false, false);
        acc1[t] = __builtin_amdgcn_wmma_f32_16x16x32_f16(false, A1, false, Bm, (short)0, acc1[t], false, false);
      }
    }
  }
}

// bias + relu + f16 store for both tiles
__device__ __forceinline__ void store2_relu(const v8f* acc0, const v8f* acc1,
                                            _Float16* __restrict__ hdst0,
                                            _Float16* __restrict__ hdst1,
                                            const float* __restrict__ b, int lane)
{
  const int m  = lane & 15;
  const int hh = lane >> 4;
#pragma unroll
  for (int t = 0; t < 8; ++t) {
    float bv = b[t * 16 + m];
#pragma unroll
    for (int r = 0; r < 8; ++r) {
      float v0 = fmaxf(acc0[t][r] + bv, 0.0f);
      float v1 = fmaxf(acc1[t][r] + bv, 0.0f);
      hdst0[(r + 8 * hh) * HSTRIDE + t * 16 + m] = (_Float16)v0;
      hdst1[(r + 8 * hh) * HSTRIDE + t * 16 + m] = (_Float16)v1;
    }
  }
}

// ---------------------------------------------------------------------------
// Main fused kernel: 256 threads = 8 wave32, 256 points per block.
// LDS: 132 KB weights + 96 KB activations + 3 KB raw = ~231 KB (< 320 KB/WGP)
// ---------------------------------------------------------------------------
__global__ void __launch_bounds__(256)
vox_mlp_kernel(const float* __restrict__ x, const float* __restrict__ grid,
               const _Float16* __restrict__ wpk,
               const float* __restrict__ b0, const float* __restrict__ b1,
               const float* __restrict__ b2, const float* __restrict__ b3,
               const float* __restrict__ bo, float* __restrict__ out)
{
  __shared__ _Float16 sW[W_TOTAL];            // all weights, f16 col-major
  __shared__ _Float16 hbuf[256 * HSTRIDE];    // activations: [0,128)=h, [128,191)=feats
  __shared__ float    rawbuf[256 * 3];

  const int tid  = threadIdx.x;
  const int lane = tid & 31;
  const int wave = tid >> 5;
  const int p    = blockIdx.x * 256 + tid;

  // ---- stage all weights into LDS via the Tensor Data Mover (wave 0) ------
  // 1-D descriptor: 16896 elements x 8 bytes = 135168 B, type=image(2),
  // count=1 (valid), no gather/iterate/pad, workgroup_mask=0 (not in cluster).
  if (wave == 0) {
    unsigned ldsa = (unsigned)(uintptr_t)(&sW[0]);      // generic LDS addr low 32b = LDS offset
    unsigned long long ga = (unsigned long long)(uintptr_t)wpk;
    u32x4 g0 = { 1u,                                    // count=1, is_restore=0, gather=0
                 ldsa,                                  // lds_addr
                 (unsigned)ga,                          // global_addr[31:0]
                 ((unsigned)(ga >> 32) & 0x01FFFFFFu) | 0x80000000u };  // addr[56:32] | type=2
    i32x8 g1 = { (int)(3u << 16),                       // data_size=3 (8B), mask=0, flags=0
                 (int)0x42000000u,                      // tensor_dim0[15:0]=16896 in [31:16]
                 0,                                     // tensor_dim0[31:16]=0, tensor_dim1 lo=0
                 (int)0x42000000u,                      // tile_dim0=16896 in [31:16]
                 0,                                     // tile_dim1=0, tile_dim2=0
                 16896,                                 // tensor_dim0_stride[31:0]
                 0, 0 };
    i32x4 z4 = {0, 0, 0, 0};
    i32x8 z8 = {0, 0, 0, 0, 0, 0, 0, 0};
    __builtin_amdgcn_tensor_load_to_lds(g0, g1, z4, z4, z8, 0);
    __builtin_amdgcn_s_wait_tensorcnt(0);
  }

  // ---- per-point preprocessing: trilinear (value + central-diff grad) ----
  float px = x[3 * p + 0], py = x[3 * p + 1], pz = x[3 * p + 2];
  float fx = (px + 1.0f) * 127.5f;
  float fy = (py + 1.0f) * 127.5f;
  float fz = (pz + 1.0f) * 127.5f;
  float x0f = floorf(fx), y0f = floorf(fy), z0f = floorf(fz);
  float xd = fx - x0f, yd = fy - y0f, zd = fz - z0f;
  int i0 = min(max((int)x0f, 0), GRES - 1), i1 = min(max((int)x0f + 1, 0), GRES - 1);
  int j0 = min(max((int)y0f, 0), GRES - 1), j1 = min(max((int)y0f + 1, 0), GRES - 1);
  int k0 = min(max((int)z0f, 0), GRES - 1), k1 = min(max((int)z0f + 1, 0), GRES - 1);

  float4 c000 = corner4(grid, i0, j0, k0), c100 = corner4(grid, i1, j0, k0);
  float4 c010 = corner4(grid, i0, j1, k0), c110 = corner4(grid, i1, j1, k0);
  float4 c001 = corner4(grid, i0, j0, k1), c101 = corner4(grid, i1, j0, k1);
  float4 c011 = corner4(grid, i0, j1, k1), c111 = corner4(grid, i1, j1, k1);
  float4 c00 = lerp4(c000, c100, xd), c10 = lerp4(c010, c110, xd);
  float4 c01 = lerp4(c001, c101, xd), c11 = lerp4(c011, c111, xd);
  float4 c0  = lerp4(c00, c10, yd),   c1  = lerp4(c01, c11, yd);
  float4 ret = lerp4(c0, c1, zd);     // (value, gx, gy, gz)

  out[p] = ret.x;                               // output 0: interpolated value
  out[NPTS + 3 * p + 0] = ret.y;                // output 1: interpolated grad
  out[NPTS + 3 * p + 1] = ret.z;
  out[NPTS + 3 * p + 2] = ret.w;

  // ---- positional encoding -> LDS feature columns [128,192) of own row ----
  {
    _Float16* frow = &hbuf[tid * HSTRIDE + 128];
    frow[0] = (_Float16)px; frow[1] = (_Float16)py; frow[2] = (_Float16)pz;
    float sc = 1.0f;
#pragma unroll
    for (int b = 0; b < 10; ++b) {
      float sx = px * sc, sy = py * sc, sz = pz * sc;
      frow[3 + 6 * b + 0] = (_Float16)__sinf(sx);
      frow[3 + 6 * b + 1] = (_Float16)__sinf(sy);
      frow[3 + 6 * b + 2] = (_Float16)__sinf(sz);
      frow[3 + 6 * b + 3] = (_Float16)__cosf(sx);
      frow[3 + 6 * b + 4] = (_Float16)__cosf(sy);
      frow[3 + 6 * b + 5] = (_Float16)__cosf(sz);
      sc *= 2.0f;
    }
    frow[63] = (_Float16)0.0f;   // pad col: must be finite (0 * NaN -> NaN in WMMA)
  }
  __syncthreads();   // weights staged (TDM complete); activation rows are wave-local

  // ---- MLP: each wave runs its two 16-point tiles jointly with WMMA ----
  const int m  = lane & 15;
  const int hh = lane >> 4;
  const int row0 = wave * 32;            // tile0 rows, tile1 = row0+16
  const _Float16* hrow0 = &hbuf[(row0 + m) * HSTRIDE];
  const _Float16* hrow1 = &hbuf[(row0 + 16 + m) * HSTRIDE];
  _Float16* hdst0 = &hbuf[row0 * HSTRIDE];
  _Float16* hdst1 = &hbuf[(row0 + 16) * HSTRIDE];

  v8f acc0[8], acc1[8];
  // L0: feats(63->64) @ W0 -> 128, relu         (A from cols 128..191)
  joint_layer<2, 8>(hrow0, hrow1, 128, sW + OFF_W0, 64, lane, acc0, acc1);
  store2_relu(acc0, acc1, hdst0, hdst1, b0, lane);
  // L1: 128 @ W1 -> 128, relu (in place; all reads precede stores, DS in-order)
  joint_layer<4, 8>(hrow0, hrow1, 0, sW + OFF_W1, 128, lane, acc0, acc1);
  store2_relu(acc0, acc1, hdst0, hdst1, b1, lane);
  // L2: 128 @ W2 -> 128, relu
  joint_layer<4, 8>(hrow0, hrow1, 0, sW + OFF_W2, 128, lane, acc0, acc1);
  store2_relu(acc0, acc1, hdst0, hdst1, b2, lane);
  // L3: concat(h,feats)=191->192 @ W3 -> 128, relu
  joint_layer<6, 8>(hrow0, hrow1, 0, sW + OFF_W3, 192, lane, acc0, acc1);
  store2_relu(acc0, acc1, hdst0, hdst1, b3, lane);
  // L4: 128 @ WO -> 3 (padded to 16), no relu
  {
    v8f ao0[1], ao1[1];
    joint_layer<4, 1>(hrow0, hrow1, 0, sW + OFF_WO, 128, lane, ao0, ao1);
    if (m < 3) {
      float bv = bo[m];
#pragma unroll
      for (int r = 0; r < 8; ++r) {
        rawbuf[(row0 + r + 8 * hh) * 3 + m]      = ao0[0][r] + bv;
        rawbuf[(row0 + 16 + r + 8 * hh) * 3 + m] = ao1[0][r] + bv;
      }
    }
  }
  __syncthreads();

  // ---- Rodrigues rotation of grad about MLP-output axis ----
  float rx = rawbuf[tid * 3 + 0], ry = rawbuf[tid * 3 + 1], rz = rawbuf[tid * 3 + 2];
  float th = sqrtf(rx * rx + ry * ry + rz * rz + 1e-12f);
  float ex = rx / th, ey = ry / th, ez = rz / th;
  float gx = ret.y, gy = ret.z, gz = ret.w;
  float an = sqrtf(gx * gx + gy * gy + gz * gz + 1e-12f);
  float vx = gx / an, vy = gy / an, vz = gz / an;
  float ct = cosf(th), st = sinf(th);
  float crx = ey * vz - ez * vy;
  float cry = ez * vx - ex * vz;
  float crz = ex * vy - ey * vx;
  float ev = ex * vx + ey * vy + ez * vz;
  float kk = (1.0f - ct) * ev;
  out[4 * NPTS + 3 * p + 0] = an * (ct * vx + st * crx + kk * ex);
  out[4 * NPTS + 3 * p + 1] = an * (ct * vy + st * cry + kk * ey);
  out[4 * NPTS + 3 * p + 2] = an * (ct * vz + st * crz + kk * ez);
}

// ---------------------------------------------------------------------------
extern "C" void kernel_launch(void* const* d_in, const int* in_sizes, int n_in,
                              void* d_out, int out_size, void* d_ws, size_t ws_size,
                              hipStream_t stream) {
  const float* x    = (const float*)d_in[0];
  const float* grid = (const float*)d_in[1];
  const float* w0   = (const float*)d_in[2];
  const float* b0   = (const float*)d_in[3];
  const float* w1   = (const float*)d_in[4];
  const float* b1   = (const float*)d_in[5];
  const float* w2   = (const float*)d_in[6];
  const float* b2   = (const float*)d_in[7];
  const float* w3   = (const float*)d_in[8];
  const float* b3   = (const float*)d_in[9];
  const float* wo   = (const float*)d_in[10];
  const float* bo   = (const float*)d_in[11];
  _Float16* wpk = (_Float16*)d_ws;
  float* out = (float*)d_out;

  vox_prep_weights<<<(W_TOTAL + 255) / 256, 256, 0, stream>>>(w0, w1, w2, w3, wo, wpk);
  vox_mlp_kernel<<<NPTS / 256, 256, 0, stream>>>(x, grid, wpk, b0, b1, b2, b3, bo, out);
}
